// FastWeightAttention_52106543235580
// MI455X (gfx1250) — matrix operand order repaired and checked by reading, not compiled, not run on hardware
//
#include <hip/hip_runtime.h>
#include <hip/hip_bf16.h>

// ---------------------------------------------------------------------------
// CDNA5 (gfx1250) implementation: all GEMMs via v_wmma_f32_16x16x32_bf16,
// wave32, 8-wave workgroups, LDS-tiled with 128-bit staging,
// Newton-Schulz fully resident in LDS.
// ---------------------------------------------------------------------------

typedef __attribute__((ext_vector_type(16))) __bf16 vbf16;
typedef __attribute__((ext_vector_type(8)))  __bf16 vbf8;
typedef __attribute__((ext_vector_type(8)))  float  vf8;

union Frag { vbf16 v; vbf8 h[2]; __bf16 e[16]; };

__device__ __forceinline__ vf8 wmma_bf16(const Frag& a, const Frag& b, vf8 c) {
  // 8-arg probe-confirmed signature: (neg_a, A, neg_b, B, c_mod, C, reuse_a, reuse_b)
  return __builtin_amdgcn_wmma_f32_16x16x32_bf16(false, a.v, false, b.v, (short)0, c, false, false);
}

// ---------------------------------------------------------------------------
// Generic tiled WMMA GEMM:  C[M x Nc] = alpha * op(A)[M x K] * op(B)[K x Nc]
//   TA: A physically stored (K x M) row-major (i.e. use A^T)
//   TB: B physically stored (Nc x K) row-major (i.e. use B^T)
//   EPI: 0=store, 1=sigmoid-store, 2=accumulate (C += )
//   WB16: also write bf16 copy of C (same geometry)
// 8 waves (256 thr): wave grid 4x2, each wave 32x32 = 2x2 WMMA fragments.
// Staging uses 128-bit global loads whenever stride/bounds allow.
// ---------------------------------------------------------------------------
template<bool TA, bool TB, int EPI, bool WB16>
__global__ __launch_bounds__(256) void k_gemm(
    const __bf16* __restrict__ Ag, const __bf16* __restrict__ Bg,
    float* Cg, __bf16* Cbg,
    int M, int Nc, int K, int lda, int ldb, int ldc,
    long long sA, long long sB, long long sC, float alpha)
{
  constexpr int BM = 128, BN = 64, BK = 32, LP = 40; // LP: padded LDS row (bf16)
  __shared__ __align__(16) __bf16 As[BM * LP];
  __shared__ __align__(16) __bf16 Bs[BN * LP];   // stored transposed: Bs[n][k]
  const int tid = threadIdx.x;
  const int bM = blockIdx.y * BM, bN = blockIdx.x * BN;
  const long long z = blockIdx.z;
  Ag += z * sA; Bg += z * sB; Cg += z * sC;
  if (WB16) Cbg += z * sC;
  const int wid = tid >> 5, lane = tid & 31;
  const int hf = lane >> 4, l16 = lane & 15;
  const int wm = wid >> 1, wn = wid & 1;
  const bool vecA = ((lda & 7) == 0);
  const bool vecB = ((ldb & 7) == 0);
  vf8 acc[2][2];
#pragma unroll
  for (int i = 0; i < 2; ++i)
#pragma unroll
    for (int j = 0; j < 2; ++j) acc[i][j] = vf8{0,0,0,0,0,0,0,0};

  for (int k0 = 0; k0 < K; k0 += BK) {
    // ---- stage A tile (BM x BK) ----
    if (!TA) {
      // contiguous along K: 8-elem (16B) chunks
      for (int c = tid; c < BM * (BK / 8); c += 256) {
        int r = c >> 2, kc = (c & 3) * 8;
        int gm = bM + r, gk = k0 + kc;
        if (vecA && gm < M && gk + 8 <= K) {
          *(vbf8*)&As[r * LP + kc] = *(const vbf8*)&Ag[(size_t)gm * lda + gk];
        } else {
#pragma unroll
          for (int u = 0; u < 8; ++u) {
            __bf16 val = (__bf16)0.0f;
            if (gm < M && gk + u < K) val = Ag[(size_t)gm * lda + gk + u];
            As[r * LP + kc + u] = val;
          }
        }
      }
    } else {
      // physical (K x M): contiguous along M -> 16B load, scatter to LDS rows
      for (int c = tid; c < BK * (BM / 8); c += 256) {
        int kk = c >> 4, mc = (c & 15) * 8;
        int gk = k0 + kk, gm0 = bM + mc;
        if (vecA && gk < K && gm0 + 8 <= M) {
          vbf8 val = *(const vbf8*)&Ag[(size_t)gk * lda + gm0];
#pragma unroll
          for (int u = 0; u < 8; ++u) As[(mc + u) * LP + kk] = val[u];
        } else {
#pragma unroll
          for (int u = 0; u < 8; ++u) {
            __bf16 val = (__bf16)0.0f;
            if (gk < K && gm0 + u < M) val = Ag[(size_t)gk * lda + gm0 + u];
            As[(mc + u) * LP + kk] = val;
          }
        }
      }
    }
    // ---- stage B tile transposed (Bs[n][k]) ----
    if (TB) {
      // physical (Nc x K): contiguous along K
      for (int c = tid; c < BN * (BK / 8); c += 256) {
        int n = c >> 2, kc = (c & 3) * 8;
        int gn = bN + n, gk = k0 + kc;
        if (vecB && gn < Nc && gk + 8 <= K) {
          *(vbf8*)&Bs[n * LP + kc] = *(const vbf8*)&Bg[(size_t)gn * ldb + gk];
        } else {
#pragma unroll
          for (int u = 0; u < 8; ++u) {
            __bf16 val = (__bf16)0.0f;
            if (gn < Nc && gk + u < K) val = Bg[(size_t)gn * ldb + gk + u];
            Bs[n * LP + kc + u] = val;
          }
        }
      }
    } else {
      // physical (K x Nc): contiguous along N -> 16B load, scatter to LDS rows
      for (int c = tid; c < BK * (BN / 8); c += 256) {
        int kk = c >> 3, nc = (c & 7) * 8;
        int gk = k0 + kk, gn0 = bN + nc;
        if (vecB && gk < K && gn0 + 8 <= Nc) {
          vbf8 val = *(const vbf8*)&Bg[(size_t)gk * ldb + gn0];
#pragma unroll
          for (int u = 0; u < 8; ++u) Bs[(nc + u) * LP + kk] = val[u];
        } else {
#pragma unroll
          for (int u = 0; u < 8; ++u) {
            __bf16 val = (__bf16)0.0f;
            if (gk < K && gn0 + u < Nc) val = Bg[(size_t)gk * ldb + gn0 + u];
            Bs[(nc + u) * LP + kk] = val;
          }
        }
      }
    }
    // prefetch hint for next K tile (speculative; dropped if OOB)
    if (k0 + BK < K) {
      size_t pa = TA ? ((size_t)(k0 + BK) * lda + bM)
                     : ((size_t)bM * lda + k0 + BK);
      __builtin_prefetch(Ag + pa + tid * 8, 0, 1);
    }
    __syncthreads();

    Frag a[2], b[2];
#pragma unroll
    for (int fm = 0; fm < 2; ++fm) {
      int ra = wm * 32 + fm * 16 + l16;
      a[fm].h[0] = *(const vbf8*)&As[ra * LP + hf * 8];
      a[fm].h[1] = *(const vbf8*)&As[ra * LP + 16 + hf * 8];
    }
#pragma unroll
    for (int fn = 0; fn < 2; ++fn) {
      int rb = wn * 32 + fn * 16 + l16;
      b[fn].h[0] = *(const vbf8*)&Bs[rb * LP + hf * 8];
      b[fn].h[1] = *(const vbf8*)&Bs[rb * LP + 16 + hf * 8];
    }
#pragma unroll
    for (int fm = 0; fm < 2; ++fm)
#pragma unroll
      for (int fn = 0; fn < 2; ++fn)
        acc[fm][fn] = wmma_bf16(a[fm], b[fn], acc[fm][fn]);
    __syncthreads();
  }

  // ---- epilogue ----
#pragma unroll
  for (int fm = 0; fm < 2; ++fm)
#pragma unroll
    for (int fn = 0; fn < 2; ++fn)
#pragma unroll
      for (int rr = 0; rr < 8; ++rr) {
        int row = bM + wm * 32 + fm * 16 + hf * 8 + rr;
        int col = bN + wn * 32 + fn * 16 + l16;
        if (row < M && col < Nc) {
          float v = alpha * acc[fm][fn][rr];
          if (EPI == 1) v = 1.0f / (1.0f + __expf(-v));
          size_t idx = (size_t)row * ldc + col;
          if (EPI == 2) Cg[idx] += v; else Cg[idx] = v;
          if (WB16) Cbg[idx] = (__bf16)v;
        }
      }
}

// ---------------------------------------------------------------------------
// Newton-Schulz(5) per head, fully LDS-resident. X is 64x1024 (bf16, padded),
// double buffered; A/Bm are 64x64. One workgroup per head (~294 KB LDS).
// Input G (r x c): if r>c then X=G^T else X=G; output written back in (r,c).
// ---------------------------------------------------------------------------
__global__ __launch_bounds__(256) void k_ns(const float* __restrict__ G,
                                            float* __restrict__ Out,
                                            int r, int c)
{
  constexpr int XP = 1032;     // padded X row (bf16 elems), 2064B = 16B-multiple
  extern __shared__ __align__(16) char smem[];
  __bf16* X0  = (__bf16*)smem;
  __bf16* X1  = X0 + 64 * XP;
  float*  Af  = (float*)(X1 + 64 * XP);     // 64 x 68 f32
  __bf16* Ab  = (__bf16*)(Af + 64 * 68);    // 64 x 72 bf16
  __bf16* Bmb = Ab + 64 * 72;               // 64 x 72 bf16
  float*  red = (float*)(Bmb + 64 * 72);    // 256 f32

  const int tid = threadIdx.x;
  const int wid = tid >> 5, lane = tid & 31, hf = lane >> 4, l16 = lane & 15;
  const size_t hoff = (size_t)blockIdx.x * r * c;
  const float* g = G + hoff;
  const bool tr = (r > c);

  // Frobenius norm
  float s = 0.0f;
  for (int i = tid; i < r * c; i += 256) { float v = g[i]; s += v * v; }
  red[tid] = s; __syncthreads();
  for (int o = 128; o > 0; o >>= 1) { if (tid < o) red[tid] += red[tid + o]; __syncthreads(); }
  const float inv = 1.0f / (sqrtf(red[0]) + 1e-7f);
  __syncthreads();

  for (int i = tid; i < 64 * 1024; i += 256) {
    int e = i >> 10, d = i & 1023;
    float v = tr ? g[(size_t)d * 64 + e] : g[(size_t)e * 1024 + d];
    X0[e * XP + d] = (__bf16)(v * inv);
  }
  __syncthreads();

  __bf16* cur = X0; __bf16* nxt = X1;
  const float aC = 3.4445f, bC = -4.775f, cC = 2.0315f;

  for (int it = 0; it < 5; ++it) {
    // ---- A = X * X^T  (64x64, K=1024); 16 tiles, 2 per wave ----
    for (int t = 0; t < 2; ++t) {
      int tt = wid * 2 + t, tm = tt >> 2, tn = tt & 3;
      vf8 acc = vf8{0,0,0,0,0,0,0,0};
      int ra = tm * 16 + l16, rb = tn * 16 + l16;
      for (int k0 = 0; k0 < 1024; k0 += 32) {
        Frag a, b;
        a.h[0] = *(const vbf8*)&cur[ra * XP + k0 + hf * 8];
        a.h[1] = *(const vbf8*)&cur[ra * XP + k0 + 16 + hf * 8];
        b.h[0] = *(const vbf8*)&cur[rb * XP + k0 + hf * 8];
        b.h[1] = *(const vbf8*)&cur[rb * XP + k0 + 16 + hf * 8];
        acc = wmma_bf16(a, b, acc);
      }
#pragma unroll
      for (int rr = 0; rr < 8; ++rr) {
        int row = tm * 16 + hf * 8 + rr, col = tn * 16 + l16;
        Af[row * 68 + col] = acc[rr];
        Ab[row * 72 + col] = (__bf16)acc[rr];
      }
    }
    __syncthreads();
    // ---- Bm = b*A + c*A*A  (A is symmetric so B-frags read Ab rows) ----
    for (int t = 0; t < 2; ++t) {
      int tt = wid * 2 + t, tm = tt >> 2, tn = tt & 3;
      vf8 acc = vf8{0,0,0,0,0,0,0,0};
      int ra = tm * 16 + l16, rb = tn * 16 + l16;
      for (int k0 = 0; k0 < 64; k0 += 32) {
        Frag a, b;
        a.h[0] = *(const vbf8*)&Ab[ra * 72 + k0 + hf * 8];
        a.h[1] = *(const vbf8*)&Ab[ra * 72 + k0 + 16 + hf * 8];
        b.h[0] = *(const vbf8*)&Ab[rb * 72 + k0 + hf * 8];
        b.h[1] = *(const vbf8*)&Ab[rb * 72 + k0 + 16 + hf * 8];
        acc = wmma_bf16(a, b, acc);
      }
#pragma unroll
      for (int rr = 0; rr < 8; ++rr) {
        int row = tm * 16 + hf * 8 + rr, col = tn * 16 + l16;
        Bmb[row * 72 + col] = (__bf16)(bC * Af[row * 68 + col] + cC * acc[rr]);
      }
    }
    __syncthreads();
    // ---- X' = a*X + Bm*X  (64x1024); 256 tiles, 32 per wave ----
    for (int t = 0; t < 32; ++t) {
      int tt = wid * 32 + t, tm = tt >> 6, tn = tt & 63;
      vf8 acc = vf8{0,0,0,0,0,0,0,0};
      int ra = tm * 16 + l16, nb = tn * 16 + l16;
      for (int k0 = 0; k0 < 64; k0 += 32) {
        Frag a, b;
        a.h[0] = *(const vbf8*)&Bmb[ra * 72 + k0 + hf * 8];
        a.h[1] = *(const vbf8*)&Bmb[ra * 72 + k0 + 16 + hf * 8];
#pragma unroll
        for (int e2 = 0; e2 < 16; ++e2)
          b.e[e2] = cur[(k0 + (e2 & 8) * 2 + hf * 8 + (e2 & 7)) * XP + nb];
        acc = wmma_bf16(a, b, acc);
      }
#pragma unroll
      for (int rr = 0; rr < 8; ++rr) {
        int row = tm * 16 + hf * 8 + rr, col = tn * 16 + l16;
        float xo = (float)cur[row * XP + col];
        nxt[row * XP + col] = (__bf16)(aC * xo + acc[rr]);
      }
    }
    __syncthreads();
    __bf16* tmp = cur; cur = nxt; nxt = tmp;
  }

  float* o = Out + hoff;
  for (int i = tid; i < 64 * 1024; i += 256) {
    int e = i >> 10, d = i & 1023;
    float v = (float)cur[e * XP + d];
    if (tr) o[(size_t)d * 64 + e] = v; else o[(size_t)e * 1024 + d] = v;
  }
}

// ---------------------------------------------------------------------------
// Elementwise / reduction kernels
// ---------------------------------------------------------------------------
__global__ __launch_bounds__(256) void k_rmsnorm(const float* __restrict__ t,
    const float* __restrict__ w, float* __restrict__ x, __bf16* __restrict__ xb, int d)
{
  int i = blockIdx.x, tid = threadIdx.x;
  const float* row = t + (size_t)i * d;
  __shared__ float red[256];
  float s = 0.0f;
  for (int j = tid; j < d; j += 256) { float v = row[j]; s += v * v; }
  red[tid] = s; __syncthreads();
  for (int o = 128; o > 0; o >>= 1) { if (tid < o) red[tid] += red[tid + o]; __syncthreads(); }
  float inv = rsqrtf(red[0] / d + 1.1920929e-7f);
  for (int j = tid; j < d; j += 256) {
    float v = row[j] * inv * w[j];
    x[(size_t)i * d + j] = v; xb[(size_t)i * d + j] = (__bf16)v;
  }
}

__global__ __launch_bounds__(256) void k_softmax_causal(float* S, __bf16* Sb, int n)
{
  int i = blockIdx.x, h = blockIdx.y, tid = threadIdx.x;
  float* s = S + ((size_t)h * n + i) * n;
  __bf16* sb = Sb + ((size_t)h * n + i) * n;
  __shared__ float red[256];
  float m = -3.402823466e38f;
  for (int j = tid; j <= i; j += 256) m = fmaxf(m, s[j]);
  red[tid] = m; __syncthreads();
  for (int o = 128; o > 0; o >>= 1) { if (tid < o) red[tid] = fmaxf(red[tid], red[tid + o]); __syncthreads(); }
  m = red[0]; __syncthreads();
  float su = 0.0f;
  for (int j = tid; j <= i; j += 256) su += __expf(s[j] - m);
  red[tid] = su; __syncthreads();
  for (int o = 128; o > 0; o >>= 1) { if (tid < o) red[tid] += red[tid + o]; __syncthreads(); }
  float inv = 1.0f / red[0];
  for (int j = tid; j < n; j += 256) {
    float v = (j <= i) ? __expf(s[j] - m) * inv : 0.0f;
    s[j] = v; sb[j] = (__bf16)v;
  }
}

__global__ __launch_bounds__(256) void k_ln_err(const float* __restrict__ tv,
    const float* __restrict__ pred, float* __restrict__ err, __bf16* __restrict__ errb, int d)
{
  int i = blockIdx.x, tid = threadIdx.x;
  const float* row = tv + (size_t)i * d;
  __shared__ float r0[256], r1[256];
  float a = 0.0f, b = 0.0f;
  for (int j = tid; j < d; j += 256) { float v = row[j]; a += v; b += v * v; }
  r0[tid] = a; r1[tid] = b; __syncthreads();
  for (int o = 128; o > 0; o >>= 1) { if (tid < o) { r0[tid] += r0[tid + o]; r1[tid] += r1[tid + o]; } __syncthreads(); }
  float mean = r0[0] / d, var = r1[0] / d - mean * mean;
  float inv = rsqrtf(var + 1e-5f);
  for (int j = tid; j < d; j += 256) {
    float v = (row[j] - mean) * inv - pred[(size_t)i * d + j];
    err[(size_t)i * d + j] = v; errb[(size_t)i * d + j] = (__bf16)v;
  }
}

__global__ __launch_bounds__(256) void k_lrs(const float* __restrict__ x,
    const float* __restrict__ wlr, float* __restrict__ lr, float* __restrict__ mlr, int d)
{
  int i = blockIdx.x, tid = threadIdx.x;
  const float* row = x + (size_t)i * d;
  __shared__ float r0[256], r1[256];
  float a = 0.0f, b = 0.0f;
  for (int j = tid; j < d; j += 256) { float v = row[j]; a += v * wlr[j]; b += v * wlr[d + j]; }
  r0[tid] = a; r1[tid] = b; __syncthreads();
  for (int o = 128; o > 0; o >>= 1) { if (tid < o) { r0[tid] += r0[tid + o]; r1[tid] += r1[tid + o]; } __syncthreads(); }
  if (tid == 0) {
    lr[i]  = 0.01f / (1.0f + __expf(-r0[0]));
    mlr[i] = 0.1f  / (1.0f + __expf(-r1[0]));
  }
}

__global__ __launch_bounds__(64) void k_bwd1(const float* __restrict__ dout,
    const float* __restrict__ gates, const float* __restrict__ out,
    const float* __restrict__ opg, __bf16* __restrict__ dub,
    __bf16* __restrict__ dgb, float* __restrict__ delta, int n, int nm1, int dh)
{
  int i = blockIdx.x, h = blockIdx.y, e = threadIdx.x;
  size_t full = ((size_t)h * n + i) * dh + e;
  size_t part = ((size_t)h * nm1 + i) * dh + e;
  float dO = dout[part], g = gates[full], o = out[full], po = opg[full];
  dub[part] = (__bf16)(dO * g);
  dgb[part] = (__bf16)(dO * po * g * (1.0f - g));
  __shared__ float red[64];
  red[e] = dO * o; __syncthreads();
  for (int off = 32; off > 0; off >>= 1) { if (e < off) red[e] += red[e + off]; __syncthreads(); }
  if (e == 0) delta[(size_t)h * nm1 + i] = red[0];
}

__global__ __launch_bounds__(256) void k_dscore(const float* __restrict__ attn,
    const float* __restrict__ dattn, const float* __restrict__ delta,
    __bf16* __restrict__ ds, int n, int nm1, int h)
{
  size_t total = (size_t)h * nm1 * nm1;
  for (size_t idx = (size_t)blockIdx.x * blockDim.x + threadIdx.x; idx < total;
       idx += (size_t)gridDim.x * blockDim.x) {
    size_t hh = idx / ((size_t)nm1 * nm1), r = idx % ((size_t)nm1 * nm1);
    int i = (int)(r / nm1), j = (int)(r % nm1);
    float a = attn[((size_t)hh * n + i) * n + j];
    ds[idx] = (__bf16)(0.125f * a * (dattn[idx] - delta[hh * nm1 + i]));
  }
}

__global__ __launch_bounds__(256) void k_outgate(const float* __restrict__ opg,
    const float* __restrict__ g, float* __restrict__ o, __bf16* __restrict__ ob, long long n)
{
  for (long long i = (long long)blockIdx.x * blockDim.x + threadIdx.x; i < n;
       i += (long long)gridDim.x * blockDim.x) {
    float v = opg[i] * g[i]; o[i] = v; ob[i] = (__bf16)v;
  }
}

__global__ __launch_bounds__(256) void k_rowscale(const float* __restrict__ s,
    const float* __restrict__ sc, __bf16* __restrict__ d, int rows, int cols)
{
  long long total = (long long)rows * cols;
  for (long long i = (long long)blockIdx.x * blockDim.x + threadIdx.x; i < total;
       i += (long long)gridDim.x * blockDim.x)
    d[i] = (__bf16)(s[i] * sc[i / cols]);
}

__global__ __launch_bounds__(256) void k_cast(const float* __restrict__ s,
    __bf16* __restrict__ d, long long n)
{
  for (long long i = (long long)blockIdx.x * blockDim.x + threadIdx.x; i < n;
       i += (long long)gridDim.x * blockDim.x) d[i] = (__bf16)s[i];
}

__global__ __launch_bounds__(256) void k_fill(float* p, float v, long long n)
{
  for (long long i = (long long)blockIdx.x * blockDim.x + threadIdx.x; i < n;
       i += (long long)gridDim.x * blockDim.x) p[i] = v;
}

__global__ __launch_bounds__(256) void k_add(const float* __restrict__ a,
    const float* __restrict__ b, float* __restrict__ d, long long n)
{
  for (long long i = (long long)blockIdx.x * blockDim.x + threadIdx.x; i < n;
       i += (long long)gridDim.x * blockDim.x) d[i] = a[i] + b[i];
}

// ---------------------------------------------------------------------------
// Launcher
// ---------------------------------------------------------------------------
extern "C" void kernel_launch(void* const* d_in, const int* in_sizes, int n_in,
                              void* d_out, int out_size, void* d_ws, size_t ws_size,
                              hipStream_t stream) {
  (void)in_sizes; (void)n_in; (void)out_size; (void)ws_size;
  constexpr int N_ = 2048, D_ = 1024, H_ = 8, DH_ = 64, NM1 = 2047;
  typedef long long ll;

  const float* tokens = (const float*)d_in[0];
  const float* wq_f   = (const float*)d_in[1];
  const float* wk_f   = (const float*)d_in[2];
  const float* wv_f   = (const float*)d_in[3];
  const float* wo_f   = (const float*)d_in[4];
  const float* wg_f   = (const float*)d_in[5];
  const float* rmsw   = (const float*)d_in[6];
  const float* wlr_f  = (const float*)d_in[7];
  const float* wtv_f  = (const float*)d_in[8];

  const size_t HDDH = (size_t)H_ * D_ * DH_;
  float* pred_o = (float*)d_out;
  float* wq_o = pred_o + (size_t)N_ * D_;
  float* wk_o = wq_o + HDDH;
  float* wv_o = wk_o + HDDH;
  float* wo_o = wv_o + HDDH;
  float* wg_o = wo_o + HDDH;

  char* wp = (char*)d_ws;
  auto carve = [&](size_t bytes) -> void* {
    void* p = (void*)wp; wp += (bytes + 255) & ~(size_t)255; return p;
  };
  float*  x      = (float*) carve((size_t)N_ * D_ * 4);
  __bf16* xb     = (__bf16*)carve((size_t)N_ * D_ * 2);
  __bf16* wq_b   = (__bf16*)carve(HDDH * 2);
  __bf16* wk_b   = (__bf16*)carve(HDDH * 2);
  __bf16* wv_b   = (__bf16*)carve(HDDH * 2);
  __bf16* wo_b   = (__bf16*)carve(HDDH * 2);
  __bf16* wg_b   = (__bf16*)carve(HDDH * 2);
  __bf16* wtv_b  = (__bf16*)carve((size_t)D_ * D_ * 2);
  float*  gatesf = (float*) carve((size_t)H_ * N_ * DH_ * 4);
  float*  qf     = (float*) carve((size_t)H_ * N_ * DH_ * 4);
  __bf16* qb     = (__bf16*)carve((size_t)H_ * N_ * DH_ * 2);
  float*  kf     = (float*) carve((size_t)H_ * N_ * DH_ * 4);
  __bf16* kb     = (__bf16*)carve((size_t)H_ * N_ * DH_ * 2);
  float*  vf     = (float*) carve((size_t)H_ * N_ * DH_ * 4);
  __bf16* vb     = (__bf16*)carve((size_t)H_ * N_ * DH_ * 2);
  float*  attnf  = (float*) carve((size_t)H_ * N_ * N_ * 4);
  __bf16* attnb  = (__bf16*)carve((size_t)H_ * N_ * N_ * 2);
  float*  opgf   = (float*) carve((size_t)H_ * N_ * DH_ * 4);
  float*  outf   = (float*) carve((size_t)H_ * N_ * DH_ * 4);
  __bf16* outb   = (__bf16*)carve((size_t)H_ * N_ * DH_ * 2);
  float*  tvf    = (float*) carve((size_t)NM1 * D_ * 4);
  float*  lr     = (float*) carve((size_t)NM1 * 4);
  float*  mlr    = (float*) carve((size_t)NM1 * 4);
  float*  errf   = (float*) carve((size_t)NM1 * D_ * 4);
  __bf16* errb   = (__bf16*)carve((size_t)NM1 * D_ * 2);
  float*  doutf  = (float*) carve((size_t)H_ * NM1 * DH_ * 4);
  __bf16* dub    = (__bf16*)carve((size_t)H_ * NM1 * DH_ * 2);
  __bf16* dgb    = (__bf16*)carve((size_t)H_ * NM1 * DH_ * 2);
  float*  delta  = (float*) carve((size_t)H_ * NM1 * 4);
  float*  dvf    = (float*) carve((size_t)H_ * NM1 * DH_ * 4);
  __bf16* dvb    = (__bf16*)carve((size_t)H_ * NM1 * DH_ * 2);
  float*  dattnf = (float*) carve((size_t)H_ * NM1 * NM1 * 4);
  __bf16* dsb    = (__bf16*)carve((size_t)H_ * NM1 * NM1 * 2);
  float*  dqf    = (float*) carve((size_t)H_ * NM1 * DH_ * 4);
  __bf16* dqb    = (__bf16*)carve((size_t)H_ * NM1 * DH_ * 2);
  float*  dkf    = (float*) carve((size_t)H_ * NM1 * DH_ * 4);
  __bf16* dkb    = (__bf16*)carve((size_t)H_ * NM1 * DH_ * 2);
  __bf16* xlrb   = (__bf16*)carve((size_t)NM1 * D_ * 2);
  __bf16* xmlrb  = (__bf16*)carve((size_t)NM1 * D_ * 2);
  __bf16* omlrb  = (__bf16*)carve((size_t)H_ * NM1 * DH_ * 2);
  float*  dwq    = (float*) carve(HDDH * 4);
  float*  dwk    = (float*) carve(HDDH * 4);
  float*  dwvr   = (float*) carve(HDDH * 4);
  float*  dwor   = (float*) carve(HDDH * 4);
  float*  dwgr   = (float*) carve(HDDH * 4);
  float*  nsv    = (float*) carve(HDDH * 4);
  float*  nso    = (float*) carve(HDDH * 4);
  float*  nsg    = (float*) carve(HDDH * 4);

  // ---- input weight casts ----
  k_cast<<<512, 256, 0, stream>>>(wq_f, wq_b, (ll)HDDH);
  k_cast<<<512, 256, 0, stream>>>(wk_f, wk_b, (ll)HDDH);
  k_cast<<<512, 256, 0, stream>>>(wv_f, wv_b, (ll)HDDH);
  k_cast<<<512, 256, 0, stream>>>(wo_f, wo_b, (ll)HDDH);
  k_cast<<<512, 256, 0, stream>>>(wg_f, wg_b, (ll)HDDH);
  k_cast<<<1024, 256, 0, stream>>>(wtv_f, wtv_b, (ll)D_ * D_);

  // ---- RMSNorm ----
  k_rmsnorm<<<N_, 256, 0, stream>>>(tokens, rmsw, x, xb, D_);

  // ---- projections: gates(sigmoid), q, k, v ----
  dim3 gproj(1, (N_ + 127) / 128, H_);
  k_gemm<false,false,1,false><<<gproj, 256, 0, stream>>>(xb, wg_b, gatesf, nullptr,
      N_, DH_, D_, D_, DH_, DH_, 0LL, (ll)D_ * DH_, (ll)N_ * DH_, 1.0f);
  k_gemm<false,false,0,true><<<gproj, 256, 0, stream>>>(xb, wq_b, qf, qb,
      N_, DH_, D_, D_, DH_, DH_, 0LL, (ll)D_ * DH_, (ll)N_ * DH_, 1.0f);
  k_gemm<false,false,0,true><<<gproj, 256, 0, stream>>>(xb, wk_b, kf, kb,
      N_, DH_, D_, D_, DH_, DH_, 0LL, (ll)D_ * DH_, (ll)N_ * DH_, 1.0f);
  k_gemm<false,false,0,true><<<gproj, 256, 0, stream>>>(xb, wv_b, vf, vb,
      N_, DH_, D_, D_, DH_, DH_, 0LL, (ll)D_ * DH_, (ll)N_ * DH_, 1.0f);

  // ---- score = q k^T * SCALE ; causal softmax (in-place) ----
  dim3 gsc((N_ + 63) / 64, (N_ + 127) / 128, H_);
  k_gemm<false,true,0,false><<<gsc, 256, 0, stream>>>(qb, kb, attnf, nullptr,
      N_, N_, DH_, DH_, DH_, N_, (ll)N_ * DH_, (ll)N_ * DH_, (ll)N_ * N_, 0.125f);
  k_softmax_causal<<<dim3(N_, H_), 256, 0, stream>>>(attnf, attnb, N_);

  // ---- out_pg = attn @ v ; out = out_pg * gates ----
  k_gemm<false,false,0,false><<<gproj, 256, 0, stream>>>(attnb, vb, opgf, nullptr,
      N_, DH_, N_, N_, DH_, DH_, (ll)N_ * N_, (ll)N_ * DH_, (ll)N_ * DH_, 1.0f);
  k_outgate<<<2048, 256, 0, stream>>>(opgf, gatesf, outf, outb, (ll)H_ * N_ * DH_);

  // ---- pred = sum_h out_h @ wo_h ----
  k_fill<<<2048, 256, 0, stream>>>(pred_o, 0.0f, (ll)N_ * D_);
  for (int h = 0; h < H_; ++h)
    k_gemm<false,false,2,false><<<dim3((D_ + 63) / 64, (N_ + 127) / 128, 1), 256, 0, stream>>>(
        outb + (size_t)h * N_ * DH_, wo_b + (size_t)h * DH_ * D_, pred_o, nullptr,
        N_, D_, DH_, DH_, D_, D_, 0LL, 0LL, 0LL, 1.0f);

  // ---- tv = x[1:] @ w_tv^T ; layernorm ; error = ln(tv) - pred[:-1] ----
  k_gemm<false,true,0,false><<<dim3((D_ + 63) / 64, (NM1 + 127) / 128, 1), 256, 0, stream>>>(
      xb + D_, wtv_b, tvf, nullptr, NM1, D_, D_, D_, D_, D_, 0LL, 0LL, 0LL, 1.0f);
  k_lrs<<<NM1, 256, 0, stream>>>(x, wlr_f, lr, mlr, D_);
  k_ln_err<<<NM1, 256, 0, stream>>>(tvf, pred_o, errf, errb, D_);

  // ---- dout = error @ wo^T ----
  dim3 gbp(1, (NM1 + 127) / 128, H_);
  k_gemm<false,true,0,false><<<gbp, 256, 0, stream>>>(errb, wo_b, doutf, nullptr,
      NM1, DH_, D_, D_, D_, DH_, 0LL, (ll)DH_ * D_, (ll)NM1 * DH_, 1.0f);
  k_bwd1<<<dim3(NM1, H_), 64, 0, stream>>>(doutf, gatesf, outf, opgf, dub, dgb, delta, N_, NM1, DH_);

  // ---- dv = attn^T @ du ; dattn = du @ v^T ----
  k_gemm<true,false,0,true><<<gbp, 256, 0, stream>>>(attnb, dub, dvf, dvb,
      NM1, DH_, NM1, N_, DH_, DH_, (ll)N_ * N_, (ll)NM1 * DH_, (ll)NM1 * DH_, 1.0f);
  k_gemm<false,true,0,false><<<dim3((NM1 + 63) / 64, (NM1 + 127) / 128, H_), 256, 0, stream>>>(
      dub, vb, dattnf, nullptr, NM1, NM1, DH_, DH_, DH_, NM1,
      (ll)NM1 * DH_, (ll)N_ * DH_, (ll)NM1 * NM1, 1.0f);
  k_dscore<<<8192, 256, 0, stream>>>(attnf, dattnf, delta, dsb, N_, NM1, H_);

  // ---- dq = dscore @ k ; dk = dscore^T @ q ----
  k_gemm<false,false,0,true><<<gbp, 256, 0, stream>>>(dsb, kb, dqf, dqb,
      NM1, DH_, NM1, NM1, DH_, DH_, (ll)NM1 * NM1, (ll)N_ * DH_, (ll)NM1 * DH_, 1.0f);
  k_gemm<true,false,0,true><<<gbp, 256, 0, stream>>>(dsb, qb, dkf, dkb,
      NM1, DH_, NM1, NM1, DH_, DH_, (ll)NM1 * NM1, (ll)N_ * DH_, (ll)NM1 * DH_, 1.0f);

  // ---- lr-scaled operands ----
  k_rowscale<<<2048, 256, 0, stream>>>(x, lr, xlrb, NM1, D_);
  k_rowscale<<<2048, 256, 0, stream>>>(x, mlr, xmlrb, NM1, D_);
  for (int h = 0; h < H_; ++h)
    k_rowscale<<<512, 256, 0, stream>>>(outf + (size_t)h * N_ * DH_, mlr,
        omlrb + (size_t)h * NM1 * DH_, NM1, DH_);

  // ---- weight-gradient GEMMs ----
  dim3 gwd(1, (D_ + 127) / 128, H_);
  k_gemm<true,false,0,false><<<gwd, 256, 0, stream>>>(xlrb, dqb, dwq, nullptr,
      D_, DH_, NM1, D_, DH_, DH_, 0LL, (ll)NM1 * DH_, (ll)D_ * DH_, 1.0f);
  k_gemm<true,false,0,false><<<gwd, 256, 0, stream>>>(xlrb, dkb, dwk, nullptr,
      D_, DH_, NM1, D_, DH_, DH_, 0LL, (ll)NM1 * DH_, (ll)D_ * DH_, 1.0f);
  k_gemm<true,false,0,false><<<gwd, 256, 0, stream>>>(xmlrb, dvb, dwvr, nullptr,
      D_, DH_, NM1, D_, DH_, DH_, 0LL, (ll)NM1 * DH_, (ll)D_ * DH_, 1.0f);
  k_gemm<true,false,0,false><<<gwd, 256, 0, stream>>>(xmlrb, dgb, dwgr, nullptr,
      D_, DH_, NM1, D_, DH_, DH_, 0LL, (ll)NM1 * DH_, (ll)D_ * DH_, 1.0f);
  k_gemm<true,false,0,false><<<dim3((D_ + 63) / 64, 1, H_), 256, 0, stream>>>(
      omlrb, errb, dwor, nullptr, DH_, D_, NM1, DH_, D_, D_,
      (ll)NM1 * DH_, 0LL, (ll)DH_ * D_, 1.0f);

  // ---- Newton-Schulz(5), one workgroup per head, ~294 KB LDS each ----
  const size_t NS_LDS = (size_t)2 * 64 * 1032 * 2 + 64 * 68 * 4 + 2 * 64 * 72 * 2 + 256 * 4;
  k_ns<<<H_, 256, NS_LDS, stream>>>(dwvr, nsv, D_, DH_);
  k_ns<<<H_, 256, NS_LDS, stream>>>(dwor, nso, DH_, D_);
  k_ns<<<H_, 256, NS_LDS, stream>>>(dwgr, nsg, D_, DH_);

  // ---- final weight updates into d_out ----
  k_add<<<512, 256, 0, stream>>>(wq_f, dwq, wq_o, (ll)HDDH);
  k_add<<<512, 256, 0, stream>>>(wk_f, dwk, wk_o, (ll)HDDH);
  k_add<<<512, 256, 0, stream>>>(wv_f, nsv, wv_o, (ll)HDDH);
  k_add<<<512, 256, 0, stream>>>(wo_f, nso, wo_o, (ll)HDDH);
  k_add<<<512, 256, 0, stream>>>(wg_f, nsg, wg_o, (ll)HDDH);
}
